// GAT_Layer_60129542495
// MI455X (gfx1250) — matrix operand order repaired and checked by reading, not compiled, not run on hardware
//
#include <hip/hip_runtime.h>
#include <hip/hip_bf16.h>
#include <cstdint>
#include <cstddef>

// ---------------------------------------------------------------------------
// GAT layer on gfx1250 (MI455X).  B=8, N=2048, F_IN=128, F_OUT=64.
// Heavy math on v_wmma_f32_16x16x32_f16; softmax stats in fp32; single
// non-temporal pass over the 134MB adjacency (the roofline-limiting stream).
// ---------------------------------------------------------------------------

typedef __attribute__((ext_vector_type(16))) _Float16 v16h;
typedef __attribute__((ext_vector_type(8)))  float    v8f;
typedef __attribute__((ext_vector_type(4)))  float    v4f;
typedef __attribute__((ext_vector_type(4)))  int      v4i;

typedef __attribute__((address_space(1))) v4i* v4i_g;   // global int4*
typedef __attribute__((address_space(3))) v4i* v4i_l;   // LDS int4*

#define B_   8
#define N_   2048
#define FIN  128
#define FOUT 64
#define ALPHA 0.2f
#define K1S  136          // padded f16 LDS stride (16B aligned, low bank conflict)

#if defined(__has_builtin)
#  if __has_builtin(__builtin_amdgcn_global_load_async_to_lds_b128)
#    define HAVE_ASYNC_B128 1
#  endif
#  if __has_builtin(__builtin_amdgcn_s_wait_asynccnt)
#    define HAVE_WAIT_ASYNC 1
#  endif
#endif

union V16U { v16h h; uint4 u[2]; };

static __device__ __forceinline__ v16h ld_v16h(const void* p) {
  V16U t;
  t.u[0] = ((const uint4*)p)[0];
  t.u[1] = ((const uint4*)p)[1];
  return t.h;
}

// ---------------------------------------------------------------------------
// k1: Wh = h @ W + b   (M=16384, K=128, N=64) via WMMA.
// Also emits WhT[b][o][j] = (f16)Wh[b][j][o], staged through LDS so the
// transposed global store is coalesced (256B rows).
// ---------------------------------------------------------------------------
__global__ void __launch_bounds__(256) k1_linear(
    const float* __restrict__ h, const float* __restrict__ W,
    const float* __restrict__ bias, float* __restrict__ Wh,
    _Float16* __restrict__ WhT)
{
  __shared__ _Float16 SBUF[64 * K1S];          // reused: WT[o][k] then TR[o][jl]
  const int t   = threadIdx.x;
  const int blk = blockIdx.x;                  // 128 rows per block

  // Stage W (row-major [128][64], f32) -> WT[o*K1S + k] f16
  for (int idx = t; idx < FIN * FOUT; idx += 256) {
    const int k = idx >> 6, o = idx & 63;
    SBUF[o * K1S + k] = (_Float16)W[idx];
  }
  __syncthreads();

  const int w    = t >> 5, lane = t & 31;
  const int half = lane >> 4, lr = lane & 15;
  const int klo  = half * 8;
  const int row0 = blk * 128 + w * 16;         // wave's 16 output rows
  const int rowA = row0 + lr;                  // this lane's A row

  v8f acc[4];
#pragma unroll
  for (int ot = 0; ot < 4; ++ot) { v8f z = {}; acc[ot] = z; }

#pragma unroll
  for (int kt = 0; kt < 4; ++kt) {
    const int kb = kt * 32;
    // A fragment: lane holds row rowA, K = {klo..klo+7, 16+klo..23+klo}
    const float* hp = h + (size_t)rowA * FIN + kb + klo;
    const v4f x0 = __builtin_nontemporal_load((const v4f*)hp);
    const v4f x1 = __builtin_nontemporal_load((const v4f*)hp + 1);
    const v4f y0 = __builtin_nontemporal_load((const v4f*)(hp + 16));
    const v4f y1 = __builtin_nontemporal_load((const v4f*)(hp + 16) + 1);
    v16h af;
    af[0]=(_Float16)x0[0]; af[1]=(_Float16)x0[1]; af[2]=(_Float16)x0[2]; af[3]=(_Float16)x0[3];
    af[4]=(_Float16)x1[0]; af[5]=(_Float16)x1[1]; af[6]=(_Float16)x1[2]; af[7]=(_Float16)x1[3];
    af[8]=(_Float16)y0[0]; af[9]=(_Float16)y0[1]; af[10]=(_Float16)y0[2]; af[11]=(_Float16)y0[3];
    af[12]=(_Float16)y1[0]; af[13]=(_Float16)y1[1]; af[14]=(_Float16)y1[2]; af[15]=(_Float16)y1[3];
#pragma unroll
    for (int ot = 0; ot < 4; ++ot) {
      // B fragment: lane col = ot*16+lr, K = {klo..klo+7, 16+klo..23+klo}
      V16U bu;
      bu.u[0] = *(const uint4*)&SBUF[(ot * 16 + lr) * K1S + kb + klo];
      bu.u[1] = *(const uint4*)&SBUF[(ot * 16 + lr) * K1S + kb + 16 + klo];
      const v16h bf = bu.h;
      acc[ot] = __builtin_amdgcn_wmma_f32_16x16x32_f16(
          false, af, false, bf, (short)0, acc[ot], false, false);
    }
  }

  __syncthreads();   // all waves done reading WT; reuse SBUF as TR

  const float bvals[4] = {bias[lr], bias[16 + lr], bias[32 + lr], bias[48 + lr]};
#pragma unroll
  for (int ot = 0; ot < 4; ++ot) {
#pragma unroll
    for (int r = 0; r < 8; ++r) {
      const float v   = acc[ot][r] + bvals[ot];
      const int  orow = row0 + r + 8 * half;   // C: row = r + 8*(lane>=16)
      Wh[(size_t)orow * FOUT + ot * 16 + lr] = v;
      SBUF[(ot * 16 + lr) * K1S + (w * 16 + r + 8 * half)] = (_Float16)v;
    }
  }
  __syncthreads();

  // Coalesced transposed store: 64 (o) x 128 (j) f16 tile.
  {
    const int bb = (blk * 128) / N_;
    const int j0 = (blk * 128) % N_;
    const int o = t >> 2, seg = t & 3;                    // 32 f16 (64B) each
    const _Float16* src = &SBUF[o * K1S + seg * 32];
    _Float16* dst = WhT + ((size_t)(bb * FOUT + o) * N_ + j0 + seg * 32);
#pragma unroll
    for (int q = 0; q < 4; ++q)
      ((uint4*)dst)[q] = ((const uint4*)src)[q];
  }
}

// ---------------------------------------------------------------------------
// k2: s1[row] = Wh[row]@a[:64], s2[row] = Wh[row]@a[64:]
// ---------------------------------------------------------------------------
__global__ void __launch_bounds__(256) k2_scores(
    const float* __restrict__ Wh, const float* __restrict__ a,
    float* __restrict__ s1, float* __restrict__ s2)
{
  __shared__ float a1s[FOUT], a2s[FOUT];
  const int t = threadIdx.x;
  if (t < FOUT) { a1s[t] = a[t]; a2s[t] = a[FOUT + t]; }
  __syncthreads();
  const int row = blockIdx.x * 256 + t;
  const float* wr = Wh + (size_t)row * FOUT;
  float v1 = 0.f, v2 = 0.f;
#pragma unroll
  for (int o = 0; o < FOUT; o += 4) {
    const v4f x = *(const v4f*)(wr + o);
    v1 += x[0] * a1s[o] + x[1] * a1s[o + 1] + x[2] * a1s[o + 2] + x[3] * a1s[o + 3];
    v2 += x[0] * a2s[o] + x[1] * a2s[o + 1] + x[2] * a2s[o + 2] + x[3] * a2s[o + 3];
  }
  s1[row] = v1;
  s2[row] = v2;
}

// ---------------------------------------------------------------------------
// k2b: m1[b] = max_j s1[b, j]
// ---------------------------------------------------------------------------
__global__ void __launch_bounds__(256) k2b_max(
    const float* __restrict__ s1, float* __restrict__ m1)
{
  __shared__ float red[256];
  const int t = threadIdx.x, bb = blockIdx.x;
  float m = -3.0e38f;
  for (int idx = t; idx < N_; idx += 256) m = fmaxf(m, s1[bb * N_ + idx]);
  red[t] = m;
  __syncthreads();
  for (int s = 128; s > 0; s >>= 1) {
    if (t < s) red[t] = fmaxf(red[t], red[t + s]);
    __syncthreads();
  }
  if (t == 0) m1[bb] = red[0];
}

// ---------------------------------------------------------------------------
// k3: single-pass masked softmax-weighted aggregation.
// out[b,i,:] = elu( (1/l_i) * sum_j exp(lrelu(s1[j]+s2[i]) - c_i) * [adj>0] * Wh[b,j,:] )
// with c_i = lrelu(max_j s1[j] + s2[i]) >= true max  (safe shift, exact softmax).
// A = P tile (16 i x 32 j, f16), B = WhT tile (32 j x 16 o, f16), fp32 acc.
// adj is a pure stream (each element read once) -> non-temporal loads.
// ---------------------------------------------------------------------------
__global__ void __launch_bounds__(256) k3_attn(
    const int* __restrict__ adj, const float* __restrict__ s1,
    const float* __restrict__ s2, const float* __restrict__ m1,
    const _Float16* __restrict__ WhT, float* __restrict__ out)
{
  __shared__ float s1s[N_];
  const int t  = threadIdx.x;
  const int it = blockIdx.x, bb = blockIdx.y;

  // Stage s1[bb] (8KB) into LDS; use the CDNA5 async-to-LDS path when available.
#ifdef HAVE_ASYNC_B128
  {
    const float* gsrc = s1 + bb * N_ + t * 8;   // 8 floats = 2 x b128 per thread
    float* ldst = &s1s[t * 8];
    __builtin_amdgcn_global_load_async_to_lds_b128(
        (v4i_g)(v4i*)gsrc, (v4i_l)ldst, 0, 0);
    __builtin_amdgcn_global_load_async_to_lds_b128(
        (v4i_g)(v4i*)(gsrc + 4), (v4i_l)(ldst + 4), 0, 0);
  }
#  ifdef HAVE_WAIT_ASYNC
  __builtin_amdgcn_s_wait_asynccnt(0);
#  else
  asm volatile("s_wait_asynccnt 0x0" ::: "memory");
#  endif
#else
  for (int idx = t; idx < N_; idx += 256) s1s[idx] = s1[bb * N_ + idx];
#endif
  __syncthreads();

  const int w    = t >> 5, lane = t & 31;
  const int half = lane >> 4, lr = lane & 15;
  const int klo  = half * 8;
  const int i0   = it * 128 + w * 16;          // wave's 16 output rows
  const int rowi = i0 + lr;                    // this lane's A (= output) row

  const float s2r = s2[bb * N_ + rowi];
  const float tc  = m1[bb] + s2r;
  const float cI  = fmaxf(tc, ALPHA * tc);     // shift >= max_j e[j, rowi]

  const int* adjp = adj + (size_t)bb * N_ * N_ + rowi;       // + j*N_
  const _Float16* bptr = WhT + (size_t)bb * FOUT * N_;

  v8f acc[4];
#pragma unroll
  for (int ot = 0; ot < 4; ++ot) { v8f z = {}; acc[ot] = z; }
  float lsum = 0.f;

#pragma unroll 2
  for (int jb = 0; jb < N_; jb += 32) {
    if (jb + 32 < N_)
      __builtin_prefetch(adjp + (size_t)(jb + 32) * N_, 0, 1);
    // s1 values for this lane's K set: two aligned 32B chunks from LDS.
    const v4f sA = *(const v4f*)&s1s[jb + klo];
    const v4f sB = *(const v4f*)&s1s[jb + klo + 4];
    const v4f sC = *(const v4f*)&s1s[jb + klo + 16];
    const v4f sD = *(const v4f*)&s1s[jb + klo + 20];
    v16h af;
#pragma unroll
    for (int e = 0; e < 16; ++e) {
      const int K = (e < 8) ? (klo + e) : (16 + klo + (e - 8));
      const int j = jb + K;
      const int ad = __builtin_nontemporal_load(adjp + (size_t)j * N_); // adj[b,j,rowi]
      const float sj = (e < 4) ? sA[e & 3]
                     : (e < 8) ? sB[e & 3]
                     : (e < 12) ? sC[e & 3] : sD[e & 3];
      float tv = sj + s2r;
      tv = fmaxf(tv, ALPHA * tv);               // leakyrelu
      const float p = (ad > 0) ? __expf(tv - cI) : 0.0f;
      lsum += p;
      af[e] = (_Float16)p;
    }
#pragma unroll
    for (int ot = 0; ot < 4; ++ot) {
      V16U bu;
      bu.u[0] = *(const uint4*)(bptr + (size_t)(ot * 16 + lr) * N_ + jb + klo);
      bu.u[1] = *(const uint4*)(bptr + (size_t)(ot * 16 + lr) * N_ + jb + 16 + klo);
      const v16h bf = bu.h;
      acc[ot] = __builtin_amdgcn_wmma_f32_16x16x32_f16(
          false, af, false, bf, (short)0, acc[ot], false, false);
    }
  }

  // Row sums: lane halves cover disjoint j -> combine; lane x holds row x%16.
  const float lt = lsum + __shfl_xor(lsum, 16, 32);
#pragma unroll
  for (int r = 0; r < 8; ++r) {
    const float lrow = __shfl(lt, r + half * 8, 32);   // sum for row r+8*half
    const float inv  = 1.0f / fmaxf(lrow, 1e-30f);
    const int  orow  = i0 + r + 8 * half;
#pragma unroll
    for (int ot = 0; ot < 4; ++ot) {
      float v = acc[ot][r] * inv;
      v = (v > 0.f) ? v : (__expf(v) - 1.0f);          // ELU
      __builtin_nontemporal_store(v, &out[((size_t)bb * N_ + orow) * FOUT + ot * 16 + lr]);
    }
  }
}

// ---------------------------------------------------------------------------
extern "C" void kernel_launch(void* const* d_in, const int* in_sizes, int n_in,
                              void* d_out, int out_size, void* d_ws, size_t ws_size,
                              hipStream_t stream)
{
  (void)in_sizes; (void)n_in; (void)out_size; (void)ws_size;
  const float* h    = (const float*)d_in[0];   // [8,2048,128] f32
  const int*   adj  = (const int*)d_in[1];     // [8,2048,2048] i32
  const float* W    = (const float*)d_in[2];   // [128,64] f32
  const float* bias = (const float*)d_in[3];   // [64] f32
  const float* a    = (const float*)d_in[4];   // [128] f32
  float* out = (float*)d_out;                  // [8,2048,64] f32

  char* ws = (char*)d_ws;
  float*    Wh  = (float*)ws;                            // 4 MB
  _Float16* WhT = (_Float16*)(ws + (size_t)(4u << 20));  // 2 MB
  float*    s1  = (float*)(ws + (size_t)(6u << 20));     // 64 KB
  float*    s2  = s1 + B_ * N_;                          // 64 KB
  float*    m1  = s2 + B_ * N_;                          // 32 B

  k1_linear<<<dim3(128), dim3(256), 0, stream>>>(h, W, bias, Wh, WhT);
  k2_scores<<<dim3(64),  dim3(256), 0, stream>>>(Wh, a, s1, s2);
  k2b_max  <<<dim3(8),   dim3(256), 0, stream>>>(s1, m1);
  k3_attn  <<<dim3(16, 8), dim3(256), 0, stream>>>(adj, s1, s2, m1, WhT, out);
}
